// RelativePositionalMultiHeadAttention_37263136260158
// MI455X (gfx1250) — matrix operand order repaired and checked by reading, not compiled, not run on hardware
//
#include <hip/hip_runtime.h>

typedef __bf16 bf16_t;
typedef __attribute__((ext_vector_type(16))) __bf16 bf16x16;
typedef __attribute__((ext_vector_type(8)))  __bf16 bf16x8;
typedef __attribute__((ext_vector_type(8)))  float  f32x8;

#define NB   8      // batch
#define NH   8      // heads
#define NS   1024   // seq
#define DM   512    // d_model
#define DK   64     // d_k

// ---------------------------------------------------------------------------
// WMMA helper: D = A(16x32 bf16) * B(32x16 bf16) + C(16x16 f32)
// ---------------------------------------------------------------------------
static __device__ __forceinline__ f32x8 wmma_bf16(bf16x16 a, bf16x16 b, f32x8 c) {
  return __builtin_amdgcn_wmma_f32_16x16x32_bf16(false, a, false, b, (short)0, c,
                                                 false, false);
}

static __device__ __forceinline__ f32x8 zero8() {
  f32x8 z = {0.f, 0.f, 0.f, 0.f, 0.f, 0.f, 0.f, 0.f};
  return z;
}

// A operand (16x32, 16-bit) ISA layout: lane L holds row m=L%16; elements 0..7
// are K = 8*(L/16)+0..7, elements 8..15 are K = 16+8*(L/16)+0..7.
static __device__ __forceinline__ bf16x16
load_a_bf16(const bf16_t* base, int row0, int ld, int kbase, int lane) {
  int m = lane & 15, half = lane >> 4;
  const bf16_t* p = base + (size_t)(row0 + m) * ld + kbase + 8 * half;
  bf16x8 lo = *(const bf16x8*)(p);
  bf16x8 hi = *(const bf16x8*)(p + 16);
  bf16x16 r;
#pragma unroll
  for (int i = 0; i < 8; ++i) { r[i] = lo[i]; r[i + 8] = hi[i]; }
  return r;
}

static __device__ __forceinline__ bf16x16
load_a_f32(const float* base, int row0, int ld, int kbase, int lane) {
  int m = lane & 15, half = lane >> 4;
  const float* p = base + (size_t)(row0 + m) * ld + kbase + 8 * half;
  bf16x16 r;
#pragma unroll
  for (int i = 0; i < 8; ++i) {
    r[i]     = (__bf16)p[i];
    r[i + 8] = (__bf16)p[16 + i];
  }
  return r;
}

// B operand (32x16, 16-bit) ISA layout: lane L holds column n=L%16; elements
// 0..7 are K = 8*(L/16)+0..7, elements 8..15 are K = 16+8*(L/16)+0..7.
// `base` must be laid out [n][k] row-major.
static __device__ __forceinline__ bf16x16
load_b_bf16(const bf16_t* base, int n0, int ld, int kbase, int lane) {
  int n = lane & 15, half = lane >> 4;
  const bf16_t* p = base + (size_t)(n0 + n) * ld + kbase + 8 * half;
  bf16x8 lo = *(const bf16x8*)(p);
  bf16x8 hi = *(const bf16x8*)(p + 16);
  bf16x16 r;
#pragma unroll
  for (int i = 0; i < 8; ++i) { r[i] = lo[i]; r[i + 8] = hi[i]; }
  return r;
}

// ---------------------------------------------------------------------------
// Kernel 1: weight prep. Transpose w_q/w_k/w_v/w_p (h,512,64)->(h,64,512) bf16
// so the WMMA B operand is two contiguous 16B loads per lane; convert w_o to bf16.
// ---------------------------------------------------------------------------
__global__ void prep_weights(const float* __restrict__ wq, const float* __restrict__ wk,
                             const float* __restrict__ wv, const float* __restrict__ wp,
                             const float* __restrict__ wo,
                             bf16_t* __restrict__ WTq, bf16_t* __restrict__ WTk,
                             bf16_t* __restrict__ WTv, bf16_t* __restrict__ WTp,
                             bf16_t* __restrict__ WOb) {
  int tid = blockIdx.x * blockDim.x + threadIdx.x;
  if (tid >= NH * DM * DK) return;                 // 262144 == 512*512 too
  int h = tid / (DM * DK);
  int rem = tid - h * DM * DK;
  int j = rem / DK;                                // d_model index
  int dk = rem - j * DK;
  size_t dst = ((size_t)h * DK + dk) * DM + j;     // [h][dk][j]
  WTq[dst] = (__bf16)wq[tid];
  WTk[dst] = (__bf16)wk[tid];
  WTv[dst] = (__bf16)wv[tid];
  WTp[dst] = (__bf16)wp[tid];
  WOb[tid] = (__bf16)wo[tid];                      // [c][j], no transpose needed
}

// ---------------------------------------------------------------------------
// Kernel 2: rel[h][d][dk] = PE[d] @ w_p[h]   (d = relative distance 0..1023)
// PE generated on the fly:  PE[pos,2i]=sin(pos*f_i), PE[pos,2i+1]=cos(pos*f_i),
// f_i = 10000^(-2i/512).   One wave per 16-row d tile.
// ---------------------------------------------------------------------------
__global__ void rel_kernel(const bf16_t* __restrict__ WTp, bf16_t* __restrict__ REL) {
  int lane = threadIdx.x;
  int m = lane & 15, half = lane >> 4;
  int d0 = blockIdx.x * 16;
  float pos = (float)(d0 + m);
  const float LN1E4_OVER_D = 9.210340371976184f / 512.0f;
  for (int h = 0; h < NH; ++h) {
    f32x8 acc0 = zero8(), acc1 = zero8(), acc2 = zero8(), acc3 = zero8();
    const bf16_t* W = WTp + (size_t)h * DK * DM;
    for (int kt = 0; kt < DM / 32; ++kt) {
      bf16x16 a;
#pragma unroll
      for (int e = 0; e < 16; ++e) {
        int j = kt * 32 + ((e < 8) ? (8 * half + e) : (16 + 8 * half + (e - 8)));
        float invf = __expf(-(float)(j & ~1) * LN1E4_OVER_D);
        float th = pos * invf;
        a[e] = (__bf16)((j & 1) ? __cosf(th) : __sinf(th));
      }
      acc0 = wmma_bf16(a, load_b_bf16(W,  0, DM, kt * 32, lane), acc0);
      acc1 = wmma_bf16(a, load_b_bf16(W, 16, DM, kt * 32, lane), acc1);
      acc2 = wmma_bf16(a, load_b_bf16(W, 32, DM, kt * 32, lane), acc2);
      acc3 = wmma_bf16(a, load_b_bf16(W, 48, DM, kt * 32, lane), acc3);
    }
#pragma unroll
    for (int r = 0; r < 8; ++r) {
      int si = r + 8 * half;
      size_t o = ((size_t)h * NS + d0 + si) * DK + m;
      REL[o +  0] = (__bf16)acc0[r];
      REL[o + 16] = (__bf16)acc1[r];
      REL[o + 32] = (__bf16)acc2[r];
      REL[o + 48] = (__bf16)acc3[r];
    }
  }
}

// ---------------------------------------------------------------------------
// Kernel 3: projections.  which==0: QU=q@wq+b_u, QV=q@wq+b_v
//                         which==1: KH=k@wk
//                         which==2: VT = (v@wv)^T  stored [b,h,dk,t]
// One wave per (16-row tile, head).
// ---------------------------------------------------------------------------
__global__ void proj_kernel(const float* __restrict__ q, const float* __restrict__ k,
                            const float* __restrict__ v,
                            const float* __restrict__ bu, const float* __restrict__ bv,
                            const bf16_t* __restrict__ WTq, const bf16_t* __restrict__ WTk,
                            const bf16_t* __restrict__ WTv,
                            bf16_t* __restrict__ QU, bf16_t* __restrict__ QV,
                            bf16_t* __restrict__ KH, bf16_t* __restrict__ VT) {
  int lane = threadIdx.x;
  int rt = blockIdx.x;            // 0..511 tile over B*S rows
  int h = blockIdx.y;             // 0..7
  int which = blockIdx.z;         // 0=q, 1=k, 2=v
  const float* X = (which == 0) ? q : (which == 1) ? k : v;
  const bf16_t* W =
      ((which == 0) ? WTq : (which == 1) ? WTk : WTv) + (size_t)h * DK * DM;
  int row0 = rt * 16;
  f32x8 acc[4] = {zero8(), zero8(), zero8(), zero8()};
  for (int kt = 0; kt < DM / 32; ++kt) {
    bf16x16 a = load_a_f32(X, row0, DM, kt * 32, lane);
#pragma unroll
    for (int nt = 0; nt < 4; ++nt)
      acc[nt] = wmma_bf16(a, load_b_bf16(W, nt * 16, DM, kt * 32, lane), acc[nt]);
  }
  int m = lane & 15, half = lane >> 4;
  if (which == 0) {
#pragma unroll
    for (int nt = 0; nt < 4; ++nt) {
      float bus = bu[h * DK + nt * 16 + m];
      float bvs = bv[h * DK + nt * 16 + m];
#pragma unroll
      for (int r = 0; r < 8; ++r) {
        int row = row0 + r + 8 * half;
        int b = row >> 10, s = row & (NS - 1);
        size_t o = (((size_t)(b * NH + h) * NS + s) * DK) + nt * 16 + m;
        QU[o] = (__bf16)(acc[nt][r] + bus);
        QV[o] = (__bf16)(acc[nt][r] + bvs);
      }
    }
  } else if (which == 1) {
#pragma unroll
    for (int nt = 0; nt < 4; ++nt)
#pragma unroll
      for (int r = 0; r < 8; ++r) {
        int row = row0 + r + 8 * half;
        int b = row >> 10, s = row & (NS - 1);
        KH[(((size_t)(b * NH + h) * NS + s) * DK) + nt * 16 + m] = (__bf16)acc[nt][r];
      }
  } else {
#pragma unroll
    for (int nt = 0; nt < 4; ++nt)
#pragma unroll
      for (int r = 0; r < 8; ++r) {
        int row = row0 + r + 8 * half;
        int b = row >> 10, t = row & (NS - 1);
        VT[(((size_t)(b * NH + h) * DK + nt * 16 + m) * NS) + t] = (__bf16)acc[nt][r];
      }
  }
}

// ---------------------------------------------------------------------------
// Kernel 4: attention.  4 cooperating waves per block, one (b,h,16-query tile).
//   scores[s,t] = ((qu[s].kh[t]) + (t<=s ? qv[s].rel[s-t] : 0)) / 8
//   softmax over t (mask all-true), then O = P @ V.
// Wave split:  pass1: key tiles round-robin; pass2: two consecutive row-pairs per wave
// (stores exp back to LDS); pass3: one d_k column tile per wave.
// Positional term only for t0 <= s0 (it is identically zero for future tiles).
// ---------------------------------------------------------------------------
#define SCS 1028   // LDS row stride (floats); 1028%64=4 -> halves hit disjoint banks
#define AWAVES 4

__global__ void attn_kernel(const bf16_t* __restrict__ QU, const bf16_t* __restrict__ QV,
                            const bf16_t* __restrict__ KH, const bf16_t* __restrict__ VT,
                            const bf16_t* __restrict__ REL, bf16_t* __restrict__ O) {
  __shared__ float sc[16][SCS];
  __shared__ float rinv[16];    // 1 / row sum
  int lane = threadIdx.x & 31;
  int w = threadIdx.x >> 5;     // wave id 0..3
  int m = lane & 15, half = lane >> 4;
  int gid = blockIdx.x;         // ((b*8+h)*64 + qt)
  int qt = gid & 63;
  int bh = gid >> 6;
  int h = bh & 7;
  int b = bh >> 3;
  int s0 = qt * 16;
  const bf16_t* QUb = QU + (size_t)bh * NS * DK;
  const bf16_t* QVb = QV + (size_t)bh * NS * DK;
  const bf16_t* KHb = KH + (size_t)bh * NS * DK;
  const bf16_t* VTb = VT + (size_t)bh * DK * NS;
  const bf16_t* RELh = REL + (size_t)h * NS * DK;

  // Query operands (A layout), K split 0..31 / 32..63 of d_k (per wave copy).
  bf16x16 qa0 = load_a_bf16(QUb, s0, DK, 0, lane);
  bf16x16 qa1 = load_a_bf16(QUb, s0, DK, 32, lane);
  bf16x16 qv0 = load_a_bf16(QVb, s0, DK, 0, lane);
  bf16x16 qv1 = load_a_bf16(QVb, s0, DK, 32, lane);

  // ---- Pass 1: raw scores into LDS (key tiles strided across waves) ------
  for (int ti = w; ti < NS / 16; ti += AWAVES) {
    int t0 = ti * 16;
    // prefetch next key tile for this wave
    if (t0 + AWAVES * 16 < NS)
      __builtin_prefetch(KHb + (size_t)(t0 + AWAVES * 16 + m) * DK, 0, 1);

    f32x8 acc = zero8();
    acc = wmma_bf16(qa0, load_b_bf16(KHb, t0, DK, 0, lane), acc);
    acc = wmma_bf16(qa1, load_b_bf16(KHb, t0, DK, 32, lane), acc);

    if (t0 <= s0) {
      // positional contribution exists only for past/diagonal tiles
      int dbase = s0 - t0 - 15;         // rel distance of column j=0
      f32x8 pp[2] = {zero8(), zero8()};
#pragma unroll
      for (int jt = 0; jt < 2; ++jt) {
        int d = dbase + jt * 16 + m;    // rel row feeding B column n=m
        bf16x16 b0, b1;
        if (d >= 0 && d < NS) {
          const bf16_t* rp = RELh + (size_t)d * DK + 8 * half;
          bf16x8 l0 = *(const bf16x8*)(rp), h0 = *(const bf16x8*)(rp + 16);
          bf16x8 l1 = *(const bf16x8*)(rp + 32), h1 = *(const bf16x8*)(rp + 48);
#pragma unroll
          for (int e = 0; e < 8; ++e) { b0[e] = l0[e]; b0[e + 8] = h0[e]; b1[e] = l1[e]; b1[e + 8] = h1[e]; }
        } else {
#pragma unroll
          for (int e = 0; e < 16; ++e) { b0[e] = (__bf16)0.f; b1[e] = (__bf16)0.f; }
        }
        pp[jt] = wmma_bf16(qv0, b0, pp[jt]);
        pp[jt] = wmma_bf16(qv1, b1, pp[jt]);
      }
      // Toeplitz gather: bd[si,ti] = P[si, 15 + si - ti]
#pragma unroll
      for (int r = 0; r < 8; ++r) {
        int si = r + 8 * half;
        int j = 15 + si - m;            // 0..30
        int src = half * 16 + (j & 15);
        float g0 = __shfl(pp[0][r], src, 32);
        float g1 = __shfl(pp[1][r], src, 32);
        float bd = (j < 16) ? g0 : g1;
        sc[si][t0 + m] = (acc[r] + bd) * 0.125f;   // 1/sqrt(64)
      }
    } else {
#pragma unroll
      for (int r = 0; r < 8; ++r)
        sc[r + 8 * half][t0 + m] = acc[r] * 0.125f;
    }
  }
  __syncthreads();

  // ---- Pass 2: softmax rows; store exp(x-max) back, keep 1/sum -----------
  // wave w handles row pairs rp = 2w and 2w+1 -> rows 4w .. 4w+3.
  for (int rp = 2 * w; rp < 2 * w + 2; ++rp) {
    int row = rp * 2 + half;
    float mx = -1e30f;
    for (int c = m; c < NS; c += 16) mx = fmaxf(mx, sc[row][c]);
#pragma unroll
    for (int off = 8; off >= 1; off >>= 1) mx = fmaxf(mx, __shfl_xor(mx, off, 32));
    float s = 0.f;
    for (int c = m; c < NS; c += 16) {
      float e = __expf(sc[row][c] - mx);
      sc[row][c] = e;
      s += e;
    }
#pragma unroll
    for (int off = 8; off >= 1; off >>= 1) s += __shfl_xor(s, off, 32);
    if (m == 0) rinv[row] = 1.0f / s;
  }
  __syncthreads();

  // ---- Pass 3: O = P @ V ; wave w computes d_k column tile nt = w --------
  int nt = w;
  f32x8 oacc = zero8();
  for (int t0 = 0; t0 < NS; t0 += 32) {
    if (t0 + 32 < NS)
      __builtin_prefetch(VTb + (size_t)(nt * 16 + m) * NS + t0 + 32, 0, 1);
    bf16x16 pa;
#pragma unroll
    for (int e = 0; e < 16; ++e) {
      int kk = (e < 8) ? (8 * half + e) : (16 + 8 * half + (e - 8));
      pa[e] = (__bf16)sc[m][t0 + kk];
    }
    oacc = wmma_bf16(pa, load_b_bf16(VTb, nt * 16, NS, t0, lane), oacc);
  }
  float ri[8];
#pragma unroll
  for (int r = 0; r < 8; ++r) ri[r] = rinv[r + 8 * half];
#pragma unroll
  for (int r = 0; r < 8; ++r) {
    int si = r + 8 * half;
    size_t o = (((size_t)b * NS + s0 + si) * NH + h) * DK + nt * 16 + m;
    O[o] = (__bf16)(oacc[r] * ri[r]);   // [b][s][h][dk] == concat layout
  }
}

// ---------------------------------------------------------------------------
// Kernel 5: out = O(8192x512 bf16) @ w_o^T ; B[k][n]=wo[c][k] is contiguous.
// ---------------------------------------------------------------------------
__global__ void out_kernel(const bf16_t* __restrict__ O, const bf16_t* __restrict__ WOb,
                           float* __restrict__ out) {
  int lane = threadIdx.x;
  int row0 = blockIdx.x * 16;
  int c0 = blockIdx.y * 16;
  f32x8 acc = zero8();
  for (int kt = 0; kt < DM / 32; ++kt) {
    bf16x16 a = load_a_bf16(O, row0, DM, kt * 32, lane);
    bf16x16 bm = load_b_bf16(WOb, c0, DM, kt * 32, lane);
    acc = wmma_bf16(a, bm, acc);
  }
  int m = lane & 15, half = lane >> 4;
#pragma unroll
  for (int r = 0; r < 8; ++r)
    out[(size_t)(row0 + r + 8 * half) * DM + c0 + m] = acc[r];
}

// ---------------------------------------------------------------------------
extern "C" void kernel_launch(void* const* d_in, const int* in_sizes, int n_in,
                              void* d_out, int out_size, void* d_ws, size_t ws_size,
                              hipStream_t stream) {
  const float* q  = (const float*)d_in[0];
  const float* k  = (const float*)d_in[1];
  const float* v  = (const float*)d_in[2];
  // d_in[3] = mask, all-true in this problem -> no-op
  const float* wq = (const float*)d_in[4];
  const float* wk = (const float*)d_in[5];
  const float* wv = (const float*)d_in[6];
  const float* wp = (const float*)d_in[7];
  const float* bu = (const float*)d_in[8];
  const float* bv = (const float*)d_in[9];
  const float* wo = (const float*)d_in[10];
  float* out = (float*)d_out;

  char* ws = (char*)d_ws;
  size_t off = 0;
  auto alloc = [&](size_t bytes) -> void* {
    void* p = ws + off;
    off += (bytes + 255) & ~(size_t)255;
    return p;
  };
  bf16_t* WTq = (bf16_t*)alloc((size_t)NH * DK * DM * 2);
  bf16_t* WTk = (bf16_t*)alloc((size_t)NH * DK * DM * 2);
  bf16_t* WTv = (bf16_t*)alloc((size_t)NH * DK * DM * 2);
  bf16_t* WTp = (bf16_t*)alloc((size_t)NH * DK * DM * 2);
  bf16_t* WOb = (bf16_t*)alloc((size_t)DM * DM * 2);
  bf16_t* REL = (bf16_t*)alloc((size_t)NH * NS * DK * 2);
  bf16_t* QU  = (bf16_t*)alloc((size_t)NB * NH * NS * DK * 2);
  bf16_t* QV  = (bf16_t*)alloc((size_t)NB * NH * NS * DK * 2);
  bf16_t* KH  = (bf16_t*)alloc((size_t)NB * NH * NS * DK * 2);
  bf16_t* VT  = (bf16_t*)alloc((size_t)NB * NH * NS * DK * 2);
  bf16_t* Obf = (bf16_t*)alloc((size_t)NB * NS * NH * DK * 2);
  (void)ws_size; (void)in_sizes; (void)n_in; (void)out_size;

  prep_weights<<<dim3((NH * DM * DK + 255) / 256), dim3(256), 0, stream>>>(
      wq, wk, wv, wp, wo, WTq, WTk, WTv, WTp, WOb);
  rel_kernel<<<dim3(NS / 16), dim3(32), 0, stream>>>(WTp, REL);
  proj_kernel<<<dim3(NB * NS / 16, NH, 3), dim3(32), 0, stream>>>(
      q, k, v, bu, bv, WTq, WTk, WTv, QU, QV, KH, VT);
  attn_kernel<<<dim3(NB * NH * NS / 16), dim3(32 * AWAVES), 0, stream>>>(
      QU, QV, KH, VT, REL, Obf);
  out_kernel<<<dim3(NB * NS / 16, DM / 16), dim3(32), 0, stream>>>(Obf, WOb, out);
}
